// SemiSparseCrossAttention_996432412693
// MI455X (gfx1250) — matrix operand order repaired and checked by reading, not compiled, not run on hardware
//
#include <hip/hip_runtime.h>

// ---------------------------------------------------------------------------
// SemiSparseCrossAttention on gfx1250 (MI455X), bf16 WMMA path + TDM staging.
// E=512, H=16, D=32, R=C=384, B=4, MS_HIDDEN=16, TANH_CLIP=10.
// ---------------------------------------------------------------------------

typedef __bf16 bf16_t;
typedef __attribute__((ext_vector_type(16))) __bf16 v16bf;
typedef __attribute__((ext_vector_type(8)))  float  v8f;
typedef unsigned int u32x4 __attribute__((ext_vector_type(4)));
typedef int          i32x4 __attribute__((ext_vector_type(4)));
typedef int          i32x8 __attribute__((ext_vector_type(8)));

#define EMB 512
#define NH  16
#define HD  32
#define RR  384
#define CC  384
#define BB  4

#if defined(__has_builtin)
#if __has_builtin(__builtin_amdgcn_tensor_load_to_lds)
#define HAVE_TDM 1
#endif
#endif
#ifndef HAVE_TDM
#define HAVE_TDM 0
#endif

__device__ __forceinline__ v8f wmma_bf16(v16bf a, v16bf b, v8f c) {
  // v_wmma_f32_16x16x32_bf16  D = A(16x32) * B(32x16) + C
  return __builtin_amdgcn_wmma_f32_16x16x32_bf16(
      /*neg_a=*/false, a, /*neg_b=*/false, b,
      /*c_mod=*/(short)0, c, /*reuse_a=*/false, /*reuse_b=*/false);
}

// A fragment (16x32 bf16): lane = (m = lane&15, g = lane>>4).
// VGPR j (dword pair): element i -> K = i + 16*g
__device__ __forceinline__ v16bf load_frag_a(const bf16_t* base, int pitch, int lane) {
  int m = lane & 15, g = (lane >> 4) & 1;
  const bf16_t* p = base + m * pitch;
  v16bf f;
#pragma unroll
  for (int j = 0; j < 8; ++j) {
    int dw = 8 * g + j;   // dword index within row
    f[2 * j]     = p[2 * dw];
    f[2 * j + 1] = p[2 * dw + 1];
  }
  return f;
}

// B fragment (32x16 bf16) loaded from B^T stored row-major (16 rows of K):
// lane = (n = lane&15, g = lane>>4); element i -> K = i + 16*g.
__device__ __forceinline__ v16bf load_frag_b(const bf16_t* base, int pitch, int lane) {
  int n = lane & 15, g = (lane >> 4) & 1;
  const bf16_t* p = base + n * pitch;
  v16bf f;
#pragma unroll
  for (int j = 0; j < 8; ++j) {
    int dw = j + 8 * g;
    f[2 * j]     = p[2 * dw];
    f[2 * j + 1] = p[2 * dw + 1];
  }
  return f;
}

#if HAVE_TDM
// TDM: DMA a 384x32 bf16 tile (row-major, contiguous in global) into LDS,
// inserting 1 dword of padding after every 16 dwords (one row) so the LDS
// tile lands with pitch 34 bf16 — bank-conflict-free for the B-fragment reads.
__device__ __forceinline__ void tdm_load_k_tile(unsigned lds_addr,
                                                unsigned long long gaddr) {
  u32x4 g0;
  g0[0] = 1u;                                      // count=1 (valid user D#)
  g0[1] = lds_addr;                                // LDS byte address
  g0[2] = (unsigned)(gaddr & 0xFFFFFFFFull);       // global_addr[31:0]
  g0[3] = (unsigned)((gaddr >> 32) & 0x1FFFFFFull) // global_addr[56:32]
        | 0x80000000u;                             // type=2 ("image")
  i32x8 g1;
  g1[0] = (1 << 16)    // data_size = 2 bytes (bf16)
        | (1 << 20)    // pad_enable
        | (3 << 22);   // pad_interval code 3 = 16 dwords (one 32-bf16 row)
                       // pad_amount  code 0 = 1 dword  -> LDS pitch 34 bf16
  g1[1] = (HD & 0xFFFF) << 16;   // tensor_dim0 = 32 (low 16 in bits 63:48)
  g1[2] = (CC & 0xFFFF) << 16;   // tensor_dim0 hi = 0 | tensor_dim1 = 384
  g1[3] = (HD & 0xFFFF) << 16;   // tensor_dim1 hi = 0 | tile_dim0 = 32
  g1[4] = CC;                    // tile_dim1 = 384, tile_dim2 = 0
  g1[5] = HD;                    // tensor_dim0_stride = 32
  g1[6] = 0;
  g1[7] = 0;
  i32x4 z4 = {0, 0, 0, 0};
#if __has_include(<hip/amd_detail/amd_gfx1250_TDM.h>)
  i32x8 z8 = {0, 0, 0, 0, 0, 0, 0, 0};
  __builtin_amdgcn_tensor_load_to_lds(g0, g1, z4, z4, z8, 0);   // clang-23 form
#else
  __builtin_amdgcn_tensor_load_to_lds(g0, g1, z4, z4, 0);       // ROCm 7.2 form
#endif
}
#endif

// ---------------------------------------------------------------------------
// Generic Y = X @ W^T + bias GEMM.  M=1536 (grid.x*128), N=512 (grid.y*64),
// K=512.  STORE_MODE: 0 -> bf16 out [b][h][row][d]   (Q, K)
//                     1 -> bf16 out [b][h][d][row]   (V transposed)
//                     2 -> f32  out [row][n]         (final projection)
// ---------------------------------------------------------------------------
template <bool A_BF16, int STORE_MODE>
__global__ __launch_bounds__(256) void gemm_wmma_kernel(
    const void* __restrict__ Aptr, const float* __restrict__ W,
    const float* __restrict__ bias, void* __restrict__ Out) {
  constexpr int BM = 128, BN = 64, BK = 32, PIT = BK + 2;
  __shared__ bf16_t As[BM][PIT];
  __shared__ bf16_t Bs[BN][PIT];

  const int tid  = threadIdx.x;
  const int lane = tid & 31;
  const int wave = tid >> 5;
  const int wm   = wave & 3;   // 4 waves along M
  const int wn   = wave >> 2;  // 2 waves along N
  const int m0   = blockIdx.x * BM;
  const int n0   = blockIdx.y * BN;

  v8f acc[2][2] = {};

  for (int k0 = 0; k0 < EMB; k0 += BK) {
    // --- stage A tile 128x32 (convert f32 -> bf16 if needed) ---
    {
      int r = tid >> 1, half = tid & 1;
      if (A_BF16) {
        const bf16_t* src = (const bf16_t*)Aptr + (size_t)(m0 + r) * EMB + k0 + half * 16;
#pragma unroll
        for (int i = 0; i < 16; ++i) As[r][half * 16 + i] = src[i];
        if (k0 + BK < EMB) __builtin_prefetch(src + BK, 0, 1);
      } else {
        const float* src = (const float*)Aptr + (size_t)(m0 + r) * EMB + k0 + half * 16;
#pragma unroll
        for (int i = 0; i < 16; ++i) As[r][half * 16 + i] = (bf16_t)src[i];
        if (k0 + BK < EMB) __builtin_prefetch(src + BK, 0, 1);
      }
    }
    // --- stage W tile 64x32 (rows of W are columns of B => already B^T) ---
    {
      int r = tid >> 2, q = tid & 3;
      const float* src = W + (size_t)(n0 + r) * EMB + k0 + q * 8;
#pragma unroll
      for (int i = 0; i < 8; ++i) Bs[r][q * 8 + i] = (bf16_t)src[i];
      if (k0 + BK < EMB) __builtin_prefetch(src + BK, 0, 1);
    }
    __syncthreads();

    v16bf a0 = load_frag_a(&As[wm * 32][0],      PIT, lane);
    v16bf a1 = load_frag_a(&As[wm * 32 + 16][0], PIT, lane);
    v16bf b0 = load_frag_b(&Bs[wn * 32][0],      PIT, lane);
    v16bf b1 = load_frag_b(&Bs[wn * 32 + 16][0], PIT, lane);
    acc[0][0] = wmma_bf16(a0, b0, acc[0][0]);
    acc[0][1] = wmma_bf16(a0, b1, acc[0][1]);
    acc[1][0] = wmma_bf16(a1, b0, acc[1][0]);
    acc[1][1] = wmma_bf16(a1, b1, acc[1][1]);
    __syncthreads();
  }

  // --- store (C layout: elem e -> row = e + 8*(lane>>4), col = lane&15) ---
  const int g = lane >> 4, nlo = lane & 15;
#pragma unroll
  for (int fm = 0; fm < 2; ++fm)
#pragma unroll
    for (int fn = 0; fn < 2; ++fn)
#pragma unroll
      for (int e = 0; e < 8; ++e) {
        int row = m0 + wm * 32 + fm * 16 + e + 8 * g;
        int n   = n0 + wn * 32 + fn * 16 + nlo;
        float v = acc[fm][fn][e] + bias[n];
        if (STORE_MODE == 2) {
          ((float*)Out)[(size_t)row * EMB + n] = v;
        } else {
          int bb = row / RR, rr = row % RR;
          int h = n >> 5, d = n & 31;
          if (STORE_MODE == 0)
            ((bf16_t*)Out)[(((size_t)(bb * NH + h) * RR) + rr) * HD + d] = (bf16_t)v;
          else
            ((bf16_t*)Out)[(((size_t)(bb * NH + h) * HD) + d) * CC + rr] = (bf16_t)v;
        }
      }
}

// ---------------------------------------------------------------------------
// Fused attention: scores -> per-edge MLP -> tanh clip / beta fill ->
// softmax (no max needed: scores in [-10,10]) -> P@V -> heads (bf16).
// Grid: (B*H, R/128), 256 threads = 8 waves, each wave owns 16 Q rows.
// ---------------------------------------------------------------------------
__global__ __launch_bounds__(256) void attn_mixed_kernel(
    const bf16_t* __restrict__ qw, const bf16_t* __restrict__ kw,
    const bf16_t* __restrict__ vtw, const float* __restrict__ cost,
    const float* __restrict__ W1, const float* __restrict__ b1,
    const float* __restrict__ W2, const float* __restrict__ b2,
    const float* __restrict__ beta, bf16_t* __restrict__ heads) {
  __shared__ bf16_t Ks[CC][34];       // K rows (col, d) = B^T for Q@K^T
  __shared__ bf16_t Vts[HD][CC + 2];  // V^T rows (d, col) = B^T for P@V
  __shared__ bf16_t Ps[8][16][34];    // per-wave prob tile relayout buffer

  const int tid = threadIdx.x, lane = tid & 31, wave = tid >> 5;
  const int bh = blockIdx.x, b = bh >> 4, h = bh & 15;
  const int row0 = blockIdx.y * 128 + wave * 16;
  const int g = lane >> 4, nlo = lane & 15;

  // ---- stage K tile: 384 x 32 bf16, LDS pitch 34 via TDM row padding ----
#if HAVE_TDM
  if (wave == 0) {
    tdm_load_k_tile((unsigned)(uintptr_t)&Ks[0][0],
                    (unsigned long long)(uintptr_t)(kw + (size_t)bh * CC * HD));
    __builtin_amdgcn_s_wait_tensorcnt(0);
  }
#else
  {
    const bf16_t* kp = kw + (size_t)bh * CC * HD;
    for (int c = tid; c < CC; c += 256) {
#pragma unroll
      for (int i = 0; i < HD; ++i) Ks[c][i] = kp[(size_t)c * HD + i];
    }
  }
#endif
  // ---- stage V^T tile: 32 x 384 bf16 (pitch 386: not TDM-pad expressible) ----
  {
    const bf16_t* vp = vtw + (size_t)bh * CC * HD;
    int d = tid >> 3, seg = tid & 7;
#pragma unroll
    for (int i = 0; i < 48; ++i) {
      int c = seg * 48 + i;
      Vts[d][c] = vp[(size_t)d * CC + c];
    }
  }
  // per-head MLP parameters (uniform across block -> SGPRs)
  float w1d[16], w1c[16], b1v[16], w2v[16];
#pragma unroll
  for (int m = 0; m < 16; ++m) {
    w1d[m] = W1[(h * 2 + 0) * 16 + m];
    w1c[m] = W1[(h * 2 + 1) * 16 + m];
    b1v[m] = b1[h * 16 + m];
    w2v[m] = W2[h * 16 + m];
  }
  const float b2v = b2[h];
  const float betav = beta[h];
  __syncthreads();

  // Q A-fragment straight from global bf16 (row-major, pitch 32)
  v16bf qf = load_frag_a(qw + ((size_t)bh * RR + row0) * HD, HD, lane);

  v8f o0 = {}, o1 = {};
  float rsum[8] = {0.f, 0.f, 0.f, 0.f, 0.f, 0.f, 0.f, 0.f};
  const float inv_sqrt_d = 0.17677669529663687f;  // 1/sqrt(32)

  for (int c0 = 0; c0 < CC; c0 += 32) {
    v16bf kf0 = load_frag_b(&Ks[c0][0],      34, lane);
    v16bf kf1 = load_frag_b(&Ks[c0 + 16][0], 34, lane);
    v8f z = {};
    v8f s0 = wmma_bf16(qf, kf0, z);
    v8f s1 = wmma_bf16(qf, kf1, z);

#pragma unroll
    for (int half = 0; half < 2; ++half) {
      v8f s = half ? s1 : s0;
      const int col = c0 + half * 16 + nlo;
#pragma unroll
      for (int e = 0; e < 8; ++e) {
        int row = row0 + e + 8 * g;
        float dotv = s[e] * inv_sqrt_d;
        float cw = cost[((size_t)b * RR + row) * CC + col];
        float logit = b2v;
#pragma unroll
        for (int m = 0; m < 16; ++m) {
          float hid = fmaxf(0.f, fmaf(w1d[m], dotv, fmaf(w1c[m], cw, b1v[m])));
          logit = fmaf(w2v[m], hid, logit);
        }
        logit = fminf(fmaxf(logit, -15.f), 15.f);
        float e2 = __expf(2.f * logit);
        float th = (e2 - 1.f) / (e2 + 1.f);           // tanh(logit)
        float sc = (cw > 0.f) ? 10.f * th : betav;    // edge mask / beta fill
        float p = __expf(sc);                          // bounded: sc in [-10,10]
        rsum[e] += p;
        Ps[wave][e + 8 * g][half * 16 + nlo] = (bf16_t)p;
      }
    }
    // C-layout -> A-layout relayout through per-wave LDS tile
    asm volatile("s_wait_dscnt 0" ::: "memory");
    v16bf pf  = load_frag_a(&Ps[wave][0][0], 34, lane);
    v16bf vf0 = load_frag_b(&Vts[0][c0],  CC + 2, lane);
    v16bf vf1 = load_frag_b(&Vts[16][c0], CC + 2, lane);
    o0 = wmma_bf16(pf, vf0, o0);
    o1 = wmma_bf16(pf, vf1, o1);
  }

  // reduce exp-sums over the 16 columns held across each 16-lane half
#pragma unroll
  for (int e = 0; e < 8; ++e) {
    float s = rsum[e];
    s += __shfl_xor(s, 1);
    s += __shfl_xor(s, 2);
    s += __shfl_xor(s, 4);
    s += __shfl_xor(s, 8);
    rsum[e] = 1.f / s;
  }
  // normalize + store heads [b][row][h*32+d] as bf16 for the output GEMM
#pragma unroll
  for (int e = 0; e < 8; ++e) {
    int row = row0 + e + 8 * g;
    size_t base = ((size_t)b * RR + row) * EMB + h * HD;
    heads[base + nlo]      = (bf16_t)(o0[e] * rsum[e]);
    heads[base + 16 + nlo] = (bf16_t)(o1[e] * rsum[e]);
  }
}

// ---------------------------------------------------------------------------
extern "C" void kernel_launch(void* const* d_in, const int* in_sizes, int n_in,
                              void* d_out, int out_size, void* d_ws, size_t ws_size,
                              hipStream_t stream) {
  const float* row_emb = (const float*)d_in[0];
  const float* col_emb = (const float*)d_in[1];
  const float* cost    = (const float*)d_in[2];
  const float* Wq = (const float*)d_in[3];
  const float* bq = (const float*)d_in[4];
  const float* Wk = (const float*)d_in[5];
  const float* bk = (const float*)d_in[6];
  const float* Wv = (const float*)d_in[7];
  const float* bv = (const float*)d_in[8];
  const float* Wo = (const float*)d_in[9];
  const float* bo = (const float*)d_in[10];
  const float* W1 = (const float*)d_in[11];
  const float* b1 = (const float*)d_in[12];
  const float* W2 = (const float*)d_in[13];
  const float* b2 = (const float*)d_in[14];
  const float* beta = (const float*)d_in[15];

  const size_t perTensor = (size_t)BB * NH * RR * HD;  // 786432 bf16 elems
  bf16_t* q     = (bf16_t*)d_ws;
  bf16_t* k     = q + perTensor;
  bf16_t* vt    = k + perTensor;
  bf16_t* heads = vt + perTensor;

  dim3 gg(12, 8), tb(256);
  gemm_wmma_kernel<false, 0><<<gg, tb, 0, stream>>>(row_emb, Wq, bq, q);
  gemm_wmma_kernel<false, 0><<<gg, tb, 0, stream>>>(col_emb, Wk, bk, k);
  gemm_wmma_kernel<false, 1><<<gg, tb, 0, stream>>>(col_emb, Wv, bv, vt);

  attn_mixed_kernel<<<dim3(BB * NH, RR / 128), tb, 0, stream>>>(
      q, k, vt, cost, W1, b1, W2, b2, beta, heads);

  gemm_wmma_kernel<true, 2><<<gg, tb, 0, stream>>>(heads, Wo, bo, (float*)d_out);
}
